// CustomAttention_57251914056447
// MI455X (gfx1250) — matrix operand order repaired and checked by reading, not compiled, not run on hardware
//
#include <hip/hip_runtime.h>
#include <hip/hip_bf16.h>
#include <math.h>
#include <stddef.h>

// Problem constants (fixed by the reference): B=4, S=2048, D=2048, H=16, Dh=128.
#define S_LEN   2048
#define D_MODEL 2048
#define N_HEADS 16
#define D_HEAD  128
#define BATCH   4
#define M_ROWS  (BATCH * S_LEN)   // 8192

// LDS row pitch for staged slabs: 64B of data + 16B TDM pad = 80B.
// Bank stride = 20 -> all 16 fragment lanes hit distinct banks; stays 16B-aligned.
#define ROWP 40                    // ushorts per LDS row

// ---------------------------------------------------------------------------
// WMMA types / helpers (gfx1250, wave32)
// ---------------------------------------------------------------------------
typedef __attribute__((ext_vector_type(4)))  unsigned int v4u;
typedef __attribute__((ext_vector_type(8)))  unsigned int v8u;
typedef __attribute__((ext_vector_type(16))) __bf16      v16bf;
typedef __attribute__((ext_vector_type(8)))  float       v8f;
typedef __attribute__((ext_vector_type(4)))  unsigned int u32x4;
typedef __attribute__((ext_vector_type(8)))  int          i32x8;
typedef __attribute__((ext_vector_type(4)))  int          i32x4;

union Frag {
  v8u   u;
  v16bf b;
  v4u   q[2];
};

__device__ __forceinline__ unsigned short f2bf(float f) {
  unsigned u = __float_as_uint(f);
  u += 0x7FFFu + ((u >> 16) & 1u);   // round-to-nearest-even
  return (unsigned short)(u >> 16);
}
__device__ __forceinline__ float bf2f(unsigned short h) {
  return __uint_as_float(((unsigned)h) << 16);
}
__device__ __forceinline__ unsigned pack2(float lo, float hi) {
  return (unsigned)f2bf(lo) | ((unsigned)f2bf(hi) << 16);
}

// A-operand 16x32 bf16 frag. Per ISA: lanes 0-15 hold K {0..7,16..23} of row M=lane,
// lanes 16-31 hold K {8..15,24..31} of row M=lane-16. Two 16B loads per lane.
__device__ __forceinline__ v16bf load_a_frag(const unsigned short* row, int k0, int half) {
  Frag f;
  const v4u* p = reinterpret_cast<const v4u*>(row + k0 + half * 8);
  f.q[0] = p[0];   // K = k0 + {0..7}  (or 8..15 for upper half)
  f.q[1] = p[2];   // K = k0 + {16..23} (or 24..31)
  return f.b;
}
// B-operand 32x16 bf16 frag: lanes 0-15 hold K {0..7,16..23} of column N=lane,
// lanes 16-31 hold K {8..15,24..31} of column N=lane-16 (same order as A).
__device__ __forceinline__ v16bf load_b_frag(const unsigned short* row, int k0, int half) {
  Frag f;
  const v4u* p = reinterpret_cast<const v4u*>(row + k0 + half * 8);
  f.q[0] = p[0];
  f.q[1] = p[2];
  return f.b;
}

__device__ __forceinline__ v8f wmma_bf16(v16bf a, v16bf b, v8f c) {
  // D = A(16x32) * B(32x16) + C, fp32 accumulate
  return __builtin_amdgcn_wmma_f32_16x16x32_bf16(false, a, false, b, (short)0, c,
                                                 false, false);
}

// ---------------------------------------------------------------------------
// Tensor Data Mover: stage one 2-D slab (128 rows x 32 bf16) global -> LDS.
// D# per CDNA5 ISA ch.8: group0 = {count/type, lds_addr, global_addr},
// group1 = dims/strides.  data_size=2B; LDS pad: 4 DWORDs every 16 DWORDs
// (one 64B row) -> 80B LDS row pitch.  Groups 2/3 zero (2-D tensor).
// Row-major source with row stride = 2048 elements (true for x/W/ctx here).
// ---------------------------------------------------------------------------
__device__ __forceinline__ void tdm_load_2d(const unsigned short* gaddr,
                                            unsigned lds_off, unsigned rows_total) {
  const unsigned long long ga = (unsigned long long)(size_t)gaddr;
  u32x4 g0;
  g0[0] = 1u;                                            // count=1 (valid), user mode
  g0[1] = lds_off;                                       // lds_addr (bytes)
  g0[2] = (unsigned)ga;                                  // global_addr[31:0]
  g0[3] = (unsigned)((ga >> 32) & 0x01FFFFFFull)         // global_addr[56:32]
          | (2u << 30);                                  // type=2 ("image")
  i32x8 g1;
  g1[0] = (int)((1u << 16)                               // data_size=1 -> 2 bytes
                | (1u << 20)                             // pad_enable
                | (3u << 22)                             // pad_interval: 16 DWORDs
                | (3u << 25));                           // pad_amount: 4 DWORDs
  g1[1] = (int)((2048u & 0xFFFFu) << 16);                // tensor_dim0[15:0] = 2048
  g1[2] = (int)((rows_total & 0xFFFFu) << 16);           // dim0[31:16]=0 | dim1[15:0]
  g1[3] = (int)((rows_total >> 16) | (32u << 16));       // dim1[31:16] | tile_dim0=32
  g1[4] = (int)128u;                                     // tile_dim1=128, tile_dim2=0
  g1[5] = (int)2048u;                                    // tensor_dim0_stride = 2048
  g1[6] = 0;                                             // stride0 hi | stride1 lo
  g1[7] = 0;                                             // stride1 hi
  // 6-arg form on this toolchain: (g0, g1, g2, g3, g4, cpol)
  __builtin_amdgcn_tensor_load_to_lds(g0, g1, (i32x4)(0), (i32x4)(0), (i32x8)(0), 0);
}

// ---------------------------------------------------------------------------
// fp32 -> bf16 conversion (bandwidth trivial vs. the GEMMs)
// ---------------------------------------------------------------------------
__global__ void cvt_f32_bf16_kernel(const float* __restrict__ in,
                                    unsigned short* __restrict__ out, long n) {
  long i = (long)blockIdx.x * blockDim.x + threadIdx.x;
  if (i < n) out[i] = f2bf(in[i]);
}

// ---------------------------------------------------------------------------
// bf16 GEMM: C[m,n] = sum_k A[m,k] * W[n,k] + bias[n]   (i.e. X @ W^T + b)
// K = lda = ldw = 2048 (hardwired into the TDM descriptors).
// Workgroup = 8 waves, 128x128 C tile; double-buffered TDM staging of the A/W
// k-slabs into LDS (wave0 -> A, wave1 -> W), fragments read via ds_load_b128.
// mode 0: bf16 row-major [M,N] | mode 1: bf16 scatter -> Vt[B,H,Dh,S] | mode 2: fp32
// ---------------------------------------------------------------------------
__global__ __launch_bounds__(256) void gemm_bf16_kernel(
    const unsigned short* __restrict__ A,
    const unsigned short* __restrict__ W,
    const float* __restrict__ bias,
    void* __restrict__ out, int mode, int Mtot, int N, int K) {
  __shared__ __align__(16) unsigned short lsA[2][128 * ROWP];
  __shared__ __align__(16) unsigned short lsW[2][128 * ROWP];

  const int lane = threadIdx.x & 31;
  const int wave = threadIdx.x >> 5;
  const int wr   = wave >> 2;      // 0..1
  const int wc   = wave & 3;       // 0..3
  const int half = lane >> 4;
  const int ln   = lane & 15;

  const long row0 = (long)blockIdx.x * 128;
  const long col0 = (long)blockIdx.y * 128;
  const long roww = row0 + wr * 64;
  const long colw = col0 + wc * 32;

  v8f acc[4][2];
#pragma unroll
  for (int i = 0; i < 4; ++i)
#pragma unroll
    for (int j = 0; j < 2; ++j) acc[i][j] = (v8f){0, 0, 0, 0, 0, 0, 0, 0};

  // Prologue: stage k-chunk 0 into buffer 0.
  if (wave == 0)
    tdm_load_2d(A + (size_t)row0 * K, (unsigned)(size_t)&lsA[0][0], (unsigned)Mtot);
  else if (wave == 1)
    tdm_load_2d(W + (size_t)col0 * K, (unsigned)(size_t)&lsW[0][0], (unsigned)N);

  for (int k0 = 0; k0 < K; k0 += 32) {
    const int cur = (k0 >> 5) & 1;
    if (wave < 2) __builtin_amdgcn_s_wait_tensorcnt(0);
    __syncthreads();   // staged slabs for `cur` visible to all waves

    if (k0 + 32 < K) {  // prefetch next slab into the other buffer
      if (wave == 0)
        tdm_load_2d(A + (size_t)row0 * K + (k0 + 32),
                    (unsigned)(size_t)&lsA[cur ^ 1][0], (unsigned)Mtot);
      else if (wave == 1)
        tdm_load_2d(W + (size_t)col0 * K + (k0 + 32),
                    (unsigned)(size_t)&lsW[cur ^ 1][0], (unsigned)N);
    }

    v16bf bf[2];
#pragma unroll
    for (int j = 0; j < 2; ++j)
      bf[j] = load_b_frag(&lsW[cur][(wc * 32 + j * 16 + ln) * ROWP], 0, half);
#pragma unroll
    for (int i = 0; i < 4; ++i) {
      v16bf af = load_a_frag(&lsA[cur][(wr * 64 + i * 16 + ln) * ROWP], 0, half);
#pragma unroll
      for (int j = 0; j < 2; ++j) acc[i][j] = wmma_bf16(af, bf[j], acc[i][j]);
    }
  }

  // Epilogue. D-layout: lanes 0-15 -> col n = colw+j*16+lane, rows = VGPR r;
  // lanes 16-31 -> same cols, rows 8+r.
#pragma unroll
  for (int j = 0; j < 2; ++j) {
    const int n = (int)(colw + j * 16 + ln);
    const float bv = bias[n];
#pragma unroll
    for (int i = 0; i < 4; ++i) {
#pragma unroll
      for (int r = 0; r < 8; ++r) {
        const long m = roww + i * 16 + r + half * 8;
        const float v = acc[i][j][r] + bv;
        if (mode == 0) {
          ((unsigned short*)out)[(size_t)m * N + n] = f2bf(v);
        } else if (mode == 2) {
          ((float*)out)[(size_t)m * N + n] = v;
        } else {
          // V scatter: m = b*S+s ; n = h*Dh+dh -> Vt[((b*H+h)*Dh+dh)*S + s]
          const long b = m >> 11, s = m & (S_LEN - 1);
          const int  h = n >> 7,  dh = n & (D_HEAD - 1);
          ((unsigned short*)out)[(((size_t)b * N_HEADS + h) * D_HEAD + dh) * S_LEN + s] =
              f2bf(v);
        }
      }
    }
  }
}

// ---------------------------------------------------------------------------
// RoPE applied in place to Q and K (bf16 storage, fp32 math).
// One thread per (b,s,h,d<64) pair.
// ---------------------------------------------------------------------------
__global__ void rope_kernel(unsigned short* __restrict__ Q,
                            unsigned short* __restrict__ K) {
  const long i = (long)blockIdx.x * blockDim.x + threadIdx.x;
  if (i >= (long)BATCH * S_LEN * N_HEADS * (D_HEAD / 2)) return;
  const int d = (int)(i & 63);
  const int h = (int)((i >> 6) & (N_HEADS - 1));
  const int s = (int)((i >> 10) & (S_LEN - 1));
  const int b = (int)(i >> 21);
  const size_t base = ((size_t)(b * S_LEN + s)) * D_MODEL + h * D_HEAD + d;

  // inv_freq = 10000^(-d/64) ; ln(10000)/64 = 0.14391156831212787
  const float ang = (float)s * __expf(-0.14391156831212787f * (float)d);
  float sn, cs;
  __sincosf(ang, &sn, &cs);

  const float q1 = bf2f(Q[base]), q2 = bf2f(Q[base + 64]);
  Q[base]      = f2bf(q1 * cs - q2 * sn);
  Q[base + 64] = f2bf(q2 * cs + q1 * sn);
  const float k1 = bf2f(K[base]), k2 = bf2f(K[base + 64]);
  K[base]      = f2bf(k1 * cs - k2 * sn);
  K[base + 64] = f2bf(k2 * cs + k1 * sn);
}

// ---------------------------------------------------------------------------
// Flash attention (non-causal, online softmax). One wave per 16-query block.
// Computes S^T = K*Q^T so softmax stats are per-lane (query = lane%16) and the
// exp'd P tile repacks into the WMMA A-operand with no cross-lane traffic.
// Grid: (S/16/4, H, B), block = 128 (4 waves).
// ---------------------------------------------------------------------------
__global__ __launch_bounds__(128) void attn_kernel(
    const unsigned short* __restrict__ Q, const unsigned short* __restrict__ Km,
    const unsigned short* __restrict__ Vt, unsigned short* __restrict__ ctx) {
  const int lane = threadIdx.x & 31;
  const int wave = threadIdx.x >> 5;
  const int half = lane >> 4;
  const int ln   = lane & 15;
  const int qblk = blockIdx.x * 4 + wave;
  const int h    = blockIdx.y;
  const int b    = blockIdx.z;
  const int q0   = qblk * 16;

  const unsigned short* Qrow  = Q  + ((size_t)(b * S_LEN + q0 + ln)) * D_MODEL + h * D_HEAD;
  const unsigned short* Kbase = Km + ((size_t)b * S_LEN) * D_MODEL + h * D_HEAD;
  const unsigned short* Vbase = Vt + ((size_t)(b * N_HEADS + h)) * D_HEAD * S_LEN;

  // Q as B-operand (column n = query = lane%16), resident for the whole loop.
  v16bf qf[4];
#pragma unroll
  for (int c = 0; c < 4; ++c) qf[c] = load_b_frag(Qrow, 32 * c, half);

  v8f ctxa[8];
#pragma unroll
  for (int t = 0; t < 8; ++t) ctxa[t] = (v8f){0, 0, 0, 0, 0, 0, 0, 0};
  float m_run = -3.0e38f, l_run = 0.0f;
  const float sc = 0.08838834764831845f;  // 1/sqrt(Dh)

  for (int j = 0; j < S_LEN; j += 32) {
    // S^T tiles: s0 = keys j..j+15, s1 = keys j+16..j+31 (m=key, n=query)
    v8f s0 = (v8f){0, 0, 0, 0, 0, 0, 0, 0};
    v8f s1 = (v8f){0, 0, 0, 0, 0, 0, 0, 0};
    const unsigned short* k0r = Kbase + (size_t)(j + ln) * D_MODEL;
    const unsigned short* k1r = Kbase + (size_t)(j + 16 + ln) * D_MODEL;
#pragma unroll
    for (int c = 0; c < 4; ++c) {
      v16bf kf0 = load_a_frag(k0r, 32 * c, half);
      v16bf kf1 = load_a_frag(k1r, 32 * c, half);
      s0 = wmma_bf16(kf0, qf[c], s0);
      s1 = wmma_bf16(kf1, qf[c], s1);
    }

    // Per-lane elements all belong to query = lane%16; partner lane (^16)
    // holds the other 16 keys. Scale, row-max, exp, row-sum.
    float mx = -3.0e38f;
#pragma unroll
    for (int r = 0; r < 8; ++r) {
      s0[r] *= sc;
      s1[r] *= sc;
      mx = fmaxf(mx, fmaxf(s0[r], s1[r]));
    }
    mx = fmaxf(mx, __shfl_xor(mx, 16, 32));
    const float m_new = fmaxf(m_run, mx);
    const float corr  = __expf(m_run - m_new);
    float rs = 0.0f;
#pragma unroll
    for (int r = 0; r < 8; ++r) {
      s0[r] = __expf(s0[r] - m_new);
      s1[r] = __expf(s1[r] - m_new);
      rs += s0[r] + s1[r];
    }
    rs += __shfl_xor(rs, 16, 32);
    l_run = l_run * corr + rs;
    m_run = m_new;

    // Broadcast the per-query rescale factor to ctx D-layout rows:
    // VGPR r holds query r (lanes<16) or 8+r (lanes>=16); stats live at lane==query.
    v8f cv;
#pragma unroll
    for (int r = 0; r < 8; ++r) cv[r] = __shfl(corr, r + half * 8, 32);

    // Repack P (S^T D-layout) into A-operand (q=M, key=K): purely lane-local.
    Frag pf;
#pragma unroll
    for (int i = 0; i < 4; ++i) {
      pf.u[i]     = pack2(s0[2 * i], s0[2 * i + 1]);
      pf.u[4 + i] = pack2(s1[2 * i], s1[2 * i + 1]);
    }

    // ctx[t] = ctx[t]*corr + P(16x32) * Vt-tile(32x16), for 8 dh tiles.
#pragma unroll
    for (int t = 0; t < 8; ++t) {
      v16bf vf = load_b_frag(Vbase + (size_t)(t * 16 + ln) * S_LEN, j, half);
#pragma unroll
      for (int r = 0; r < 8; ++r) ctxa[t][r] *= cv[r];
      ctxa[t] = wmma_bf16(pf.b, vf, ctxa[t]);
    }
  }

  // Normalize by l and store ctx (bf16, [B*S, D] with col = h*Dh + dh).
  const float linv = 1.0f / l_run;
  v8f lv;
#pragma unroll
  for (int r = 0; r < 8; ++r) lv[r] = __shfl(linv, r + half * 8, 32);
#pragma unroll
  for (int t = 0; t < 8; ++t) {
    const int n = h * D_HEAD + t * 16 + ln;
#pragma unroll
    for (int r = 0; r < 8; ++r) {
      const long m = q0 + r + half * 8;
      ctx[((size_t)b * S_LEN + m) * D_MODEL + n] = f2bf(ctxa[t][r] * lv[r]);
    }
  }
}

// ---------------------------------------------------------------------------
// Orchestration. Workspace layout (bytes), total 192 MB:
//   xb @0 (32M) | wqb @32M | wkb @40M | wvb @48M | wob @56M (8M each)
//   Qb @64M | Kb @96M | Vt @128M | ctxb @160M (32M each)
// ---------------------------------------------------------------------------
extern "C" void kernel_launch(void* const* d_in, const int* in_sizes, int n_in,
                              void* d_out, int out_size, void* d_ws, size_t ws_size,
                              hipStream_t stream) {
  (void)in_sizes; (void)n_in; (void)out_size; (void)ws_size;
  const float* x  = (const float*)d_in[0];
  const float* wq = (const float*)d_in[1];
  const float* bq = (const float*)d_in[2];
  const float* wk = (const float*)d_in[3];
  const float* bk = (const float*)d_in[4];
  const float* wv = (const float*)d_in[5];
  const float* bv = (const float*)d_in[6];
  const float* wo = (const float*)d_in[7];
  const float* bo = (const float*)d_in[8];

  char* ws = (char*)d_ws;
  const size_t MB = 1024u * 1024u;
  unsigned short* xb   = (unsigned short*)(ws);
  unsigned short* wqb  = (unsigned short*)(ws + 32 * MB);
  unsigned short* wkb  = (unsigned short*)(ws + 40 * MB);
  unsigned short* wvb  = (unsigned short*)(ws + 48 * MB);
  unsigned short* wob  = (unsigned short*)(ws + 56 * MB);
  unsigned short* Qb   = (unsigned short*)(ws + 64 * MB);
  unsigned short* Kb   = (unsigned short*)(ws + 96 * MB);
  unsigned short* Vt   = (unsigned short*)(ws + 128 * MB);
  unsigned short* ctxb = (unsigned short*)(ws + 160 * MB);

  const long nx = (long)M_ROWS * D_MODEL;     // 16.8M
  const long nw = (long)D_MODEL * D_MODEL;    // 4.2M
  cvt_f32_bf16_kernel<<<(nx + 255) / 256, 256, 0, stream>>>(x, xb, nx);
  cvt_f32_bf16_kernel<<<(nw + 255) / 256, 256, 0, stream>>>(wq, wqb, nw);
  cvt_f32_bf16_kernel<<<(nw + 255) / 256, 256, 0, stream>>>(wk, wkb, nw);
  cvt_f32_bf16_kernel<<<(nw + 255) / 256, 256, 0, stream>>>(wv, wvb, nw);
  cvt_f32_bf16_kernel<<<(nw + 255) / 256, 256, 0, stream>>>(wo, wob, nw);

  dim3 gg(M_ROWS / 128, D_MODEL / 128);  // 64 x 16
  gemm_bf16_kernel<<<gg, 256, 0, stream>>>(xb, wqb, bq, Qb, 0, M_ROWS, D_MODEL, D_MODEL);
  gemm_bf16_kernel<<<gg, 256, 0, stream>>>(xb, wkb, bk, Kb, 0, M_ROWS, D_MODEL, D_MODEL);
  gemm_bf16_kernel<<<gg, 256, 0, stream>>>(xb, wvb, bv, Vt, 1, M_ROWS, D_MODEL, D_MODEL);

  const long np = (long)BATCH * S_LEN * N_HEADS * (D_HEAD / 2);  // 8.4M
  rope_kernel<<<(np + 255) / 256, 256, 0, stream>>>(Qb, Kb);

  dim3 ga(S_LEN / 16 / 4, N_HEADS, BATCH);  // 32 x 16 x 4
  attn_kernel<<<ga, 128, 0, stream>>>(Qb, Kb, Vt, ctxb);

  gemm_bf16_kernel<<<gg, 256, 0, stream>>>(ctxb, wob, bo, d_out, 2, M_ROWS, D_MODEL,
                                           D_MODEL);
}
